// NewDistanceEstimator_21990232555677
// MI455X (gfx1250) — compile-verified, matrix operand order checked by reading, not run on hardware
//
#include <hip/hip_runtime.h>
#include <hip/hip_bf16.h>

typedef __attribute__((ext_vector_type(16))) _Float16 v16h;
typedef __attribute__((ext_vector_type(8)))  float    v8f;

constexpr int Nn = 100000;   // nodes per graph
constexpr int Ee = 1600000;  // edges per graph
constexpr int Bb = 512;      // batches
constexpr float SCALE_INV = 1.0f / 281474976710655.0f;  // 1/(2^48-1)

// ---------------------------------------------------------------- utilities
__global__ void zero_f32(float* __restrict__ p, int n) {
  int i = blockIdx.x * blockDim.x + threadIdx.x;
  if (i < n) p[i] = 0.0f;
}

// ------------------------------------------------------- node encoder (f32)
// x0[n,32] = relu(norm*idW1+idb1) @ idW2 + idb2   (norm ~1e-9: keep f32)
__global__ void node_encode(const int* __restrict__ names,
                            const float* __restrict__ W1, const float* __restrict__ b1,
                            const float* __restrict__ W2, const float* __restrict__ b2,
                            float* __restrict__ x0) {
  __shared__ float sW1[32], sb1[32], sW2[1024], sb2[32];
  for (int i = threadIdx.x; i < 1024; i += blockDim.x) sW2[i] = W2[i];
  if (threadIdx.x < 32) {
    sW1[threadIdx.x] = W1[threadIdx.x];
    sb1[threadIdx.x] = b1[threadIdx.x];
    sb2[threadIdx.x] = b2[threadIdx.x];
  }
  __syncthreads();
  int n = blockIdx.x * blockDim.x + threadIdx.x;
  if (n >= Nn) return;
  float norm = ((float)names[n] + 2.0f) * SCALE_INV;
  norm = fminf(fmaxf(norm, 0.0f), 1.0f);
  float h[32];
#pragma unroll
  for (int k = 0; k < 32; ++k) h[k] = fmaxf(fmaf(norm, sW1[k], sb1[k]), 0.0f);
  float* out = x0 + (size_t)n * 32;
  for (int c = 0; c < 32; ++c) {
    float acc = sb2[c];
#pragma unroll
    for (int k = 0; k < 32; ++k) acc = fmaf(h[k], sW2[k * 32 + c], acc);
    out[c] = acc;
  }
}

// ------------------------------------------- conv1 message + scatter (C=32)
// 8 threads per edge, 4 channels each.  e computed in-register:
//   h32 = relu(a*edW1+edb1) ;  e[c] = edb2[c] + h32 @ edW2[:,c]
//   m   = relu(x0[src] + e) ; atomicAdd agg[dst]
__global__ void conv1_msg(const int* __restrict__ ei, const float* __restrict__ attr,
                          const float* __restrict__ eW1, const float* __restrict__ eb1,
                          const float* __restrict__ eW2, const float* __restrict__ eb2,
                          const float* __restrict__ x0, float* __restrict__ agg) {
  __shared__ float sW1[32], sb1[32], sW2[1024], sb2[32];
  for (int i = threadIdx.x; i < 1024; i += blockDim.x) sW2[i] = eW2[i];
  if (threadIdx.x < 32) {
    sW1[threadIdx.x] = eW1[threadIdx.x];
    sb1[threadIdx.x] = eb1[threadIdx.x];
    sb2[threadIdx.x] = eb2[threadIdx.x];
  }
  __syncthreads();
  int gid = blockIdx.x * blockDim.x + threadIdx.x;
  if (gid >= Ee * 8) return;
  int e  = gid >> 3;
  int c4 = (gid & 7) << 2;
  float a = attr[e];
  int src = ei[e];
  int dst = ei[Ee + e];
  float h[32];
#pragma unroll
  for (int k = 0; k < 32; ++k) h[k] = fmaxf(fmaf(a, sW1[k], sb1[k]), 0.0f);
  float a0 = sb2[c4 + 0], a1 = sb2[c4 + 1], a2 = sb2[c4 + 2], a3 = sb2[c4 + 3];
#pragma unroll
  for (int k = 0; k < 32; ++k) {
    float hk = h[k];
    const float* w = &sW2[k * 32 + c4];
    a0 = fmaf(hk, w[0], a0); a1 = fmaf(hk, w[1], a1);
    a2 = fmaf(hk, w[2], a2); a3 = fmaf(hk, w[3], a3);
  }
  float4 xv = *(const float4*)(x0 + (size_t)src * 32 + c4);
  float* p = agg + (size_t)dst * 32 + c4;
  atomicAdd(p + 0, fmaxf(xv.x + a0, 0.0f));
  atomicAdd(p + 1, fmaxf(xv.y + a1, 0.0f));
  atomicAdd(p + 2, fmaxf(xv.z + a2, 0.0f));
  atomicAdd(p + 3, fmaxf(xv.w + a3, 0.0f));
}

// --------------------------- fold edge-MLP layer2 through linW (for conv2)
// Wc[32,64] = edW2 @ linW ; bc[64] = edb2 @ linW + linb
__global__ void combine_lin(const float* __restrict__ eW2, const float* __restrict__ eb2,
                            const float* __restrict__ linW, const float* __restrict__ linb,
                            float* __restrict__ Wc, float* __restrict__ bc) {
  int gid = blockIdx.x * blockDim.x + threadIdx.x;
  if (gid < 2048) {
    int k = gid >> 6, c = gid & 63;
    float acc = 0.0f;
#pragma unroll
    for (int j = 0; j < 32; ++j) acc = fmaf(eW2[k * 32 + j], linW[j * 64 + c], acc);
    Wc[gid] = acc;
  } else if (gid < 2112) {
    int c = gid - 2048;
    float acc = linb[c];
#pragma unroll
    for (int j = 0; j < 32; ++j) acc = fmaf(eb2[j], linW[j * 64 + c], acc);
    bc[c] = acc;
  }
}

// ------------------------------------------- conv2 message + scatter (C=64)
// 8 threads per edge, 8 channels each.  e2 = h32 @ Wc + bc (folded edge MLP).
__global__ void conv2_msg(const int* __restrict__ ei, const float* __restrict__ attr,
                          const float* __restrict__ eW1, const float* __restrict__ eb1,
                          const float* __restrict__ Wc, const float* __restrict__ bc,
                          const float* __restrict__ x1, float* __restrict__ agg) {
  __shared__ float sW1[32], sb1[32], sWc[2048], sbc[64];
  for (int i = threadIdx.x; i < 2048; i += blockDim.x) sWc[i] = Wc[i];
  if (threadIdx.x < 64) sbc[threadIdx.x] = bc[threadIdx.x];
  if (threadIdx.x < 32) { sW1[threadIdx.x] = eW1[threadIdx.x]; sb1[threadIdx.x] = eb1[threadIdx.x]; }
  __syncthreads();
  int gid = blockIdx.x * blockDim.x + threadIdx.x;
  if (gid >= Ee * 8) return;
  int e  = gid >> 3;
  int c8 = (gid & 7) << 3;
  float a = attr[e];
  int src = ei[e];
  int dst = ei[Ee + e];
  float h[32];
#pragma unroll
  for (int k = 0; k < 32; ++k) h[k] = fmaxf(fmaf(a, sW1[k], sb1[k]), 0.0f);
  float acc[8];
#pragma unroll
  for (int j = 0; j < 8; ++j) acc[j] = sbc[c8 + j];
#pragma unroll
  for (int k = 0; k < 32; ++k) {
    float hk = h[k];
    const float* w = &sWc[k * 64 + c8];
#pragma unroll
    for (int j = 0; j < 8; ++j) acc[j] = fmaf(hk, w[j], acc[j]);
  }
  const float* xp = x1 + (size_t)src * 64 + c8;
  float4 v0 = *(const float4*)(xp);
  float4 v1 = *(const float4*)(xp + 4);
  float xr[8] = {v0.x, v0.y, v0.z, v0.w, v1.x, v1.y, v1.z, v1.w};
  float* p = agg + (size_t)dst * 64 + c8;
#pragma unroll
  for (int j = 0; j < 8; ++j) atomicAdd(p + j, fmaxf(xr[j] + acc[j], 0.0f));
}

// -------------------------------------------------- WMMA GEMM (node MLPs)
// out[M,64] = relu( (A (+A2)) [M,K] @ W[K,64] + bias ), K = 32*K32.
// One wave per 16-row tile (M % 16 == 0 so EXEC stays all-ones).
// Register layouts per CDNA5 ISA 7.12.2:
//  A 16x32 f16: lane M=l, half g: VGPR0-3 -> K[8g..8g+8), VGPR4-7 -> K[16+8g..+8)
//  B 32x16 f16: lane N=l, same K interleave (mirrors sparse-B table halved)
//  C/D f32:     lane N=l, c[r] = row (8g + r)
template <int K32, bool ADD2>
__global__ void gemm_rows_wmma(const float* __restrict__ A, const float* __restrict__ A2,
                               const float* __restrict__ W, const float* __restrict__ bias,
                               float* __restrict__ out, int M) {
  constexpr int K = 32 * K32;
  const int lane = threadIdx.x & 31;
  const int l = lane & 15;
  const int g = lane >> 4;
  const int wave = blockIdx.x * (blockDim.x >> 5) + (threadIdx.x >> 5);
  const int numWaves = gridDim.x * (blockDim.x >> 5);

  // B tiles held in registers for the whole kernel (W is 64x64 max)
  v16h Breg[4][K32];
#pragma unroll
  for (int nt = 0; nt < 4; ++nt) {
#pragma unroll
    for (int kc = 0; kc < K32; ++kc) {
      int n = nt * 16 + l;
      v16h b;
#pragma unroll
      for (int v = 0; v < 8; ++v) {
        int k0 = kc * 32 + ((v >= 4) ? 16 : 0) + g * 8 + (v & 3) * 2;
        b[2 * v + 0] = (_Float16)W[(size_t)(k0 + 0) * 64 + n];
        b[2 * v + 1] = (_Float16)W[(size_t)(k0 + 1) * 64 + n];
      }
      Breg[nt][kc] = b;
    }
  }
  float bval[4];
#pragma unroll
  for (int nt = 0; nt < 4; ++nt) bval[nt] = bias[nt * 16 + l];

  const int numTiles = M >> 4;
  for (int t = wave; t < numTiles; t += numWaves) {
    const int row = t * 16 + l;
    v16h Areg[K32];
#pragma unroll
    for (int kc = 0; kc < K32; ++kc) {
      const float* ap = A + (size_t)row * K + kc * 32;
      float4 p0 = *(const float4*)(ap + g * 8);
      float4 p1 = *(const float4*)(ap + g * 8 + 4);
      float4 q0 = *(const float4*)(ap + 16 + g * 8);
      float4 q1 = *(const float4*)(ap + 16 + g * 8 + 4);
      if (ADD2) {
        const float* bp = A2 + (size_t)row * K + kc * 32;
        float4 r0 = *(const float4*)(bp + g * 8);
        float4 r1 = *(const float4*)(bp + g * 8 + 4);
        float4 s0 = *(const float4*)(bp + 16 + g * 8);
        float4 s1 = *(const float4*)(bp + 16 + g * 8 + 4);
        p0.x += r0.x; p0.y += r0.y; p0.z += r0.z; p0.w += r0.w;
        p1.x += r1.x; p1.y += r1.y; p1.z += r1.z; p1.w += r1.w;
        q0.x += s0.x; q0.y += s0.y; q0.z += s0.z; q0.w += s0.w;
        q1.x += s1.x; q1.y += s1.y; q1.z += s1.z; q1.w += s1.w;
      }
      v16h av;
      av[0]  = (_Float16)p0.x; av[1]  = (_Float16)p0.y;
      av[2]  = (_Float16)p0.z; av[3]  = (_Float16)p0.w;
      av[4]  = (_Float16)p1.x; av[5]  = (_Float16)p1.y;
      av[6]  = (_Float16)p1.z; av[7]  = (_Float16)p1.w;
      av[8]  = (_Float16)q0.x; av[9]  = (_Float16)q0.y;
      av[10] = (_Float16)q0.z; av[11] = (_Float16)q0.w;
      av[12] = (_Float16)q1.x; av[13] = (_Float16)q1.y;
      av[14] = (_Float16)q1.z; av[15] = (_Float16)q1.w;
      Areg[kc] = av;
    }
#pragma unroll
    for (int nt = 0; nt < 4; ++nt) {
      v8f c = {};
#pragma unroll
      for (int kc = 0; kc < K32; ++kc) {
        c = __builtin_amdgcn_wmma_f32_16x16x32_f16(
            false, Areg[kc], false, Breg[nt][kc], (short)0, c, false, false);
      }
      const int rowBase = t * 16 + 8 * g;
      const float bb = bval[nt];
#pragma unroll
      for (int r = 0; r < 8; ++r) {
        float v = c[r] + bb;
        v = v > 0.0f ? v : 0.0f;
        out[(size_t)(rowBase + r) * 64 + nt * 16 + l] = v;
      }
    }
  }
}

// ------------------------------------------------------ mean-pool (scatter)
__global__ void pool_kernel(const float* __restrict__ x, const int* __restrict__ batch,
                            float* __restrict__ pool, float* __restrict__ cnt) {
  int gid = blockIdx.x * blockDim.x + threadIdx.x;
  if (gid >= Nn * 16) return;
  int n = gid >> 4, c4 = (gid & 15) << 2;
  int b = batch[n];
  float4 v = *(const float4*)(x + (size_t)n * 64 + c4);
  float* p = pool + (size_t)b * 64 + c4;
  atomicAdd(p + 0, v.x); atomicAdd(p + 1, v.y);
  atomicAdd(p + 2, v.z); atomicAdd(p + 3, v.w);
  if ((gid & 15) == 0) atomicAdd(cnt + b, 1.0f);
}

// ---------------------------------------------------------- final regressor
// z = [s | g | depth] (129) ; out[b] = relu(z@rW1+rb1) @ rW2 + rb2
__global__ void regressor(const float* __restrict__ pool_s, const float* __restrict__ cnt_s,
                          const float* __restrict__ pool_g, const float* __restrict__ cnt_g,
                          const float* __restrict__ depth,
                          const float* __restrict__ rW1, const float* __restrict__ rb1,
                          const float* __restrict__ rW2, const float* __restrict__ rb2,
                          float* __restrict__ out) {
  __shared__ float z[129];
  __shared__ float red[64];
  int b = blockIdx.x, t = threadIdx.x;  // 64 threads
  float cs = fmaxf(cnt_s[b], 1.0f), cg = fmaxf(cnt_g[b], 1.0f);
  z[t]      = pool_s[(size_t)b * 64 + t] / cs;
  z[64 + t] = pool_g[(size_t)b * 64 + t] / cg;
  if (t == 0) z[128] = depth[b];
  __syncthreads();
  float acc = rb1[t];
  for (int k = 0; k < 129; ++k) acc = fmaf(z[k], rW1[k * 64 + t], acc);
  acc = fmaxf(acc, 0.0f);
  red[t] = acc * rW2[t];
  __syncthreads();
  for (int s = 32; s > 0; s >>= 1) {
    if (t < s) red[t] += red[t + s];
    __syncthreads();
  }
  if (t == 0) out[b] = red[0] + rb2[0];
}

// ---------------------------------------------------------------- launcher
extern "C" void kernel_launch(void* const* d_in, const int* in_sizes, int n_in,
                              void* d_out, int out_size, void* d_ws, size_t ws_size,
                              hipStream_t stream) {
  (void)in_sizes; (void)n_in; (void)out_size; (void)ws_size;

  const int*   names_s = (const int*)d_in[0];
  const int*   ei_s    = (const int*)d_in[1];
  const float* attr_s  = (const float*)d_in[2];
  const int*   batch_s = (const int*)d_in[3];
  const int*   names_g = (const int*)d_in[4];
  const int*   ei_g    = (const int*)d_in[5];
  const float* attr_g  = (const float*)d_in[6];
  const int*   batch_g = (const int*)d_in[7];
  const float* depth   = (const float*)d_in[8];
  const float* idW1 = (const float*)d_in[9],  *idb1 = (const float*)d_in[10];
  const float* idW2 = (const float*)d_in[11], *idb2 = (const float*)d_in[12];
  const float* edW1 = (const float*)d_in[13], *edb1 = (const float*)d_in[14];
  const float* edW2 = (const float*)d_in[15], *edb2 = (const float*)d_in[16];
  const float* s1W1 = (const float*)d_in[17], *s1b1 = (const float*)d_in[18];
  const float* s1W2 = (const float*)d_in[19], *s1b2 = (const float*)d_in[20];
  const float* s2W1 = (const float*)d_in[21], *s2b1 = (const float*)d_in[22];
  const float* s2W2 = (const float*)d_in[23], *s2b2 = (const float*)d_in[24];
  const float* s2linW = (const float*)d_in[25], *s2linb = (const float*)d_in[26];
  const float* g1W1 = (const float*)d_in[27], *g1b1 = (const float*)d_in[28];
  const float* g1W2 = (const float*)d_in[29], *g1b2 = (const float*)d_in[30];
  const float* g2W1 = (const float*)d_in[31], *g2b1 = (const float*)d_in[32];
  const float* g2W2 = (const float*)d_in[33], *g2b2 = (const float*)d_in[34];
  const float* g2linW = (const float*)d_in[35], *g2linb = (const float*)d_in[36];
  const float* rW1 = (const float*)d_in[37], *rb1 = (const float*)d_in[38];
  const float* rW2 = (const float*)d_in[39], *rb2 = (const float*)d_in[40];
  float* outp = (float*)d_out;

  // workspace layout (floats); all offsets multiples of 64 floats (256B)
  float* ws   = (float*)d_ws;
  float* x0   = ws;                        // N*32
  float* agg  = x0  + (size_t)Nn * 32;     // N*64
  float* hbuf = agg + (size_t)Nn * 64;     // N*64
  float* xbuf = hbuf + (size_t)Nn * 64;    // N*64
  float* Wc   = xbuf + (size_t)Nn * 64;    // 2048
  float* bc   = Wc + 2048;                 // 64
  float* pool_s = bc + 64;                 // B*64
  float* cnt_s  = pool_s + (size_t)Bb * 64;// B   (contiguous after pool_s)
  float* pool_g = cnt_s + Bb;              // B*64
  float* cnt_g  = pool_g + (size_t)Bb * 64;// B

  const int TB = 256;
  const int gNodes  = (Nn + TB - 1) / TB;           // 391
  const int gEdges8 = (Ee * 8 + TB - 1) / TB;       // 50000
  const int gA32    = (Nn * 32 + TB - 1) / TB;
  const int gA64    = (Nn * 64 + TB - 1) / TB;
  const int gPool   = (Nn * 16 + TB - 1) / TB;
  const int gGemm   = 782;                          // ~1 tile per wave (6250 tiles)
  const int gZp     = (Bb * 64 + Bb + TB - 1) / TB;

  auto run_graph = [&](const int* names, const int* ei, const float* attr, const int* batch,
                       const float* c1W1, const float* c1b1, const float* c1W2, const float* c1b2,
                       const float* c2W1, const float* c2b1, const float* c2W2, const float* c2b2,
                       const float* linW, const float* linb, float* pool, float* cnt) {
    node_encode<<<gNodes, TB, 0, stream>>>(names, idW1, idb1, idW2, idb2, x0);
    // --- conv1 ---
    zero_f32<<<gA32, TB, 0, stream>>>(agg, Nn * 32);
    conv1_msg<<<gEdges8, TB, 0, stream>>>(ei, attr, edW1, edb1, edW2, edb2, x0, agg);
    gemm_rows_wmma<1, true ><<<gGemm, TB, 0, stream>>>(x0, agg, c1W1, c1b1, hbuf, Nn);
    gemm_rows_wmma<2, false><<<gGemm, TB, 0, stream>>>(hbuf, nullptr, c1W2, c1b2, xbuf, Nn);
    // --- conv2 ---
    combine_lin<<<9, TB, 0, stream>>>(edW2, edb2, linW, linb, Wc, bc);
    zero_f32<<<gA64, TB, 0, stream>>>(agg, Nn * 64);
    conv2_msg<<<gEdges8, TB, 0, stream>>>(ei, attr, edW1, edb1, Wc, bc, xbuf, agg);
    gemm_rows_wmma<2, true ><<<gGemm, TB, 0, stream>>>(xbuf, agg, c2W1, c2b1, hbuf, Nn);
    gemm_rows_wmma<2, false><<<gGemm, TB, 0, stream>>>(hbuf, nullptr, c2W2, c2b2, xbuf, Nn);
    // --- pool ---
    zero_f32<<<gZp, TB, 0, stream>>>(pool, Bb * 64 + Bb);  // pool + cnt contiguous
    pool_kernel<<<gPool, TB, 0, stream>>>(xbuf, batch, pool, cnt);
  };

  run_graph(names_s, ei_s, attr_s, batch_s,
            s1W1, s1b1, s1W2, s1b2, s2W1, s2b1, s2W2, s2b2, s2linW, s2linb,
            pool_s, cnt_s);
  run_graph(names_g, ei_g, attr_g, batch_g,
            g1W1, g1b1, g1W2, g1b2, g2W1, g2b1, g2W2, g2b2, g2linW, g2linb,
            pool_g, cnt_g);

  regressor<<<Bb, 64, 0, stream>>>(pool_s, cnt_s, pool_g, cnt_g, depth,
                                   rW1, rb1, rW2, rb2, outp);
}
